// VAEMemoryBank_43825846289093
// MI455X (gfx1250) — matrix operand order repaired and checked by reading, not compiled, not run on hardware
//
#include <hip/hip_runtime.h>
#include <hip/hip_bf16.h>

typedef __attribute__((ext_vector_type(16))) _Float16 v16h;
typedef __attribute__((ext_vector_type(8)))  float    v8f;

#define DMODEL 192
#define TSEQ   4096
#define BANK   1000
#define BANKP  1024
#define NH     2
#define DK     96
#define DKP    112    // 96 dk cols + ones column (96) + zero pad
#define TT     64     // t-rows per workgroup
#define NB     16     // batch
// (1/sqrt(96)) * log2(e): fold score scale into Q, softmax in log2 domain
#define QSCALE 0.14724445f

union AF { v16h v; unsigned u[8]; };

// ---- fragment loaders ------------------------------------------------------
// A-operand (16x32 f16): lane holds row (lane&15); VGPR v K-base:
//   lo-half: v0..3 -> K=2v, v4..7 -> K=16+2(v-4); hi-half adds +8.
__device__ __forceinline__ v16h fragA(const _Float16* p, int hi) {
  AF x;
#pragma unroll
  for (int v = 0; v < 8; ++v) {
    int kb = ((v & 4) << 2) + (hi << 3) + ((v & 3) << 1);
    x.u[v] = *reinterpret_cast<const unsigned*>(p + kb);
  }
  return x.v;
}
// B-operand (32x16 f16): lane holds column (lane&15); same K order as fragA.
__device__ __forceinline__ v16h fragB(const _Float16* p, int hi) {
  AF x;
#pragma unroll
  for (int v = 0; v < 8; ++v) {
    int kb = ((v & 4) << 2) + (hi << 3) + ((v & 3) << 1);
    x.u[v] = *reinterpret_cast<const unsigned*>(p + kb);
  }
  return x.v;
}

__device__ __forceinline__ v8f wmma16(v16h a, v16h b, v8f c) {
  return __builtin_amdgcn_wmma_f32_16x16x32_f16(false, a, false, b, (short)0, c,
                                                false, false);
}

// CDNA5 async global->LDS copy (ASYNCcnt-tracked)
__device__ __forceinline__ void async_ld_b128(unsigned lds_off,
                                              unsigned long long gaddr) {
  asm volatile("global_load_async_to_lds_b128 %0, %1, off" ::"v"(lds_off),
               "v"(gaddr)
               : "memory");
}
__device__ __forceinline__ void wait_async0() {
  asm volatile("s_wait_asynccnt 0" ::: "memory");
}

// ---- kernel A: K/V projection of memory bank (batch-independent) -----------
// kh: [NH][BANKP][DK]  f16 (B-operand for S = Q*K^T : lane=bank, K=dk)
// vT: [NH][DKP][BANKP] f16 (B-operand for O = P*V   : lane=dk,  K=bank)
//     row 96 = ones for valid banks (accumulates softmax denominator l)
__global__ __launch_bounds__(256) void kv_prep(
    const float* __restrict__ mem, const float* __restrict__ Wk,
    const float* __restrict__ bk, const float* __restrict__ Wv,
    const float* __restrict__ bv, _Float16* __restrict__ kh,
    _Float16* __restrict__ vT) {
  int gid = blockIdx.x * 256 + threadIdx.x;   // NH*DKP*BANKP threads
  int o2 = gid >> 10;
  int s = gid & (BANKP - 1);
  if (o2 >= NH * DKP) return;
  int h = o2 / DKP;
  int dkp = o2 - h * DKP;
  if (dkp < DK) {
    int o = h * DK + dkp;
    float ka = 0.f, va = 0.f;
    if (s < BANK) {
      ka = bk[o];
      va = bv[o];
      const float* wkr = Wk + o * DMODEL;
      const float* wvr = Wv + o * DMODEL;
#pragma unroll 4
      for (int c = 0; c < DMODEL; ++c) {
        float m = mem[c * BANK + s];
        ka += wkr[c] * m;
        va += wvr[c] * m;
      }
    }
    kh[((size_t)h * BANKP + s) * DK + dkp] = (_Float16)ka;
    vT[((size_t)h * DKP + dkp) * BANKP + s] = (_Float16)va;
  } else {
    float v = (dkp == DK && s < BANK) ? 1.0f : 0.0f;
    vT[((size_t)h * DKP + dkp) * BANKP + s] = (_Float16)v;
  }
}

// ---- kernel A2: one-time f32->f16 conversion of Wq / Wo --------------------
__global__ __launch_bounds__(256) void w_cvt(const float* __restrict__ Wq,
                                             const float* __restrict__ Wo,
                                             _Float16* __restrict__ wq16,
                                             _Float16* __restrict__ wo16) {
  int gid = blockIdx.x * 256 + threadIdx.x;
  if (gid < DMODEL * DMODEL) {
    wq16[gid] = (_Float16)Wq[gid];
    wo16[gid] = (_Float16)Wo[gid];
  }
}

// ---- kernel B: fused Q-proj + flash attention + O-proj ---------------------
__global__ __launch_bounds__(256) void fused_attn(
    const float* __restrict__ z, const _Float16* __restrict__ wq16,
    const float* __restrict__ bq, const _Float16* __restrict__ wo16,
    const float* __restrict__ bo, const _Float16* __restrict__ kh,
    const _Float16* __restrict__ vT, float* __restrict__ out) {
  // LDS layout (56KB):
  //  [0,24576)     qld[2][64][96] f16    (aliased as zstage float[96][64] in ph0)
  //  [24576,49152) zattn[64][192] f16    (z tile, later merged attention)
  //  [49152,57344) pld[8][16][32] f16    (per-wave P staging)
  __shared__ alignas(16) unsigned char smem[57344];
  auto qld    = reinterpret_cast<_Float16(*)[TT][DK]>(smem);
  auto zstage = reinterpret_cast<float(*)[TT]>(smem);
  auto zattn  = reinterpret_cast<_Float16(*)[DMODEL]>(smem + 24576);
  auto pld    = reinterpret_cast<_Float16(*)[16][32]>(smem + 49152);

  const int tid  = threadIdx.x;
  const int wave = tid >> 5;
  const int lane = tid & 31;
  const int lr   = lane & 15;
  const int hi   = lane >> 4;
  const int bidx = blockIdx.x >> 6;   // 64 t-tiles per batch
  const int tile = blockIdx.x & 63;
  const int t0   = tile * TT;
  const float* zb = z + (size_t)bidx * DMODEL * TSEQ;
  const unsigned sbase = (unsigned)(size_t)(void*)smem;

  // phase 0: stage z tile via async global->LDS (two 96-channel halves),
  //          then transpose+convert into zattn[t][c] f16
  for (int half = 0; half < 2; ++half) {
#pragma unroll
    for (int i = tid; i < DK * 16; i += 256) {   // 96 rows x 16 b128 quads
      int c = i >> 4, q4 = (i & 15) << 2;
      unsigned loff = sbase + (unsigned)(c * TT + q4) * 4u;
      async_ld_b128(loff, (unsigned long long)(zb +
                    (size_t)(half * DK + c) * TSEQ + t0 + q4));
    }
    wait_async0();
    __syncthreads();
    for (int i = tid; i < DK * TT; i += 256) {   // transpose 96x64
      int c = i >> 6, r = i & 63;
      zattn[r][half * DK + c] = (_Float16)zstage[c][r];
    }
    __syncthreads();
  }

  // phase 1: Q = (Wq*z + bq) * QSCALE -> qld[h][t][dk]  (48 tiles, 6/wave)
  for (int tt = wave; tt < 48; tt += 8) {
    int h = tt / 24, rem = tt - h * 24;
    int mt = rem >> 2, nt = rem & 3;
    int row = h * DK + mt * 16 + lr;
    v8f acc = {};
#pragma unroll
    for (int kc = 0; kc < 6; ++kc) {
      v16h a = fragA(wq16 + (size_t)row * DMODEL + kc * 32, hi);
      v16h b = fragB(&zattn[nt * 16 + lr][kc * 32], hi);
      acc = wmma16(a, b, acc);
    }
#pragma unroll
    for (int j = 0; j < 8; ++j) {
      int och = h * DK + mt * 16 + j + hi * 8;
      qld[h][nt * 16 + lr][mt * 16 + j + hi * 8] =
          (_Float16)((acc[j] + bq[och]) * QSCALE);
    }
  }
  __syncthreads();

  // phase 2: online-softmax attention; each wave: 16 t-rows of one head.
  // Scores already in log2 domain. No masking needed: padded K rows give
  // score 0 (shift-invariant), padded V rows are 0, ones-column is 0 on pads.
  const int h2 = wave >> 2;
  const int m0 = (wave & 3) << 4;
  v16h aq[3];
#pragma unroll
  for (int kt = 0; kt < 3; ++kt)
    aq[kt] = fragA(&qld[h2][m0 + lr][kt * 32], hi);
  const _Float16* khh = kh + (size_t)h2 * BANKP * DK;
  const _Float16* vth = vT + (size_t)h2 * DKP * BANKP;

  v8f oacc[7];   // 6 dk tiles + ones-column tile (accumulates l)
#pragma unroll
  for (int n = 0; n < 7; ++n) oacc[n] = v8f{};
  float mrow[8];
#pragma unroll
  for (int j = 0; j < 8; ++j) mrow[j] = -3.0e38f;

  for (int chk = 0; chk < BANKP / 64; ++chk) {
    int c0 = chk << 6;
    // S over 64 banks: 4 column tiles x 3 K-steps = 12 WMMA
    v8f s0 = {}, s1 = {}, s2 = {}, s3 = {};
    const _Float16* kp = khh + (size_t)(c0 + lr) * DK;
#pragma unroll
    for (int kt = 0; kt < 3; ++kt) {
      s0 = wmma16(aq[kt], fragB(kp + kt * 32, hi), s0);
      s1 = wmma16(aq[kt], fragB(kp + 16 * DK + kt * 32, hi), s1);
      s2 = wmma16(aq[kt], fragB(kp + 32 * DK + kt * 32, hi), s2);
      s3 = wmma16(aq[kt], fragB(kp + 48 * DK + kt * 32, hi), s3);
    }
    // online softmax (row max only; l comes from the ones-column of V)
    float alpha[8], e2a[8], e3a[8];
#pragma unroll
    for (int j = 0; j < 8; ++j) {
      float f0 = s0[j], f1 = s1[j], f2 = s2[j], f3 = s3[j];
      float mx = fmaxf(fmaxf(f0, f1), fmaxf(f2, f3));
#pragma unroll
      for (int d = 1; d < 16; d <<= 1) mx = fmaxf(mx, __shfl_xor(mx, d, 32));
      float mn = fmaxf(mrow[j], mx);
      alpha[j] = exp2f(mrow[j] - mn);
      mrow[j] = mn;
      pld[wave][j + hi * 8][lr]      = (_Float16)exp2f(f0 - mn);
      pld[wave][j + hi * 8][16 + lr] = (_Float16)exp2f(f1 - mn);
      e2a[j] = exp2f(f2 - mn);
      e3a[j] = exp2f(f3 - mn);
    }
#pragma unroll
    for (int n = 0; n < 7; ++n)
#pragma unroll
      for (int j = 0; j < 8; ++j) oacc[n][j] *= alpha[j];
    // P banks 0..31 -> A frag, then reuse staging buffer for banks 32..63
    // (same-wave LDS ops are in-order: loads complete before later stores)
    v16h ap0 = fragA(&pld[wave][lr][0], hi);
#pragma unroll
    for (int j = 0; j < 8; ++j) {
      pld[wave][j + hi * 8][lr]      = (_Float16)e2a[j];
      pld[wave][j + hi * 8][16 + lr] = (_Float16)e3a[j];
    }
    v16h ap1 = fragA(&pld[wave][lr][0], hi);
#pragma unroll
    for (int n = 0; n < 7; ++n) {
      const _Float16* vb = vth + (size_t)(n * 16 + lr) * BANKP + c0;
      oacc[n] = wmma16(ap0, fragB(vb, hi), oacc[n]);
      oacc[n] = wmma16(ap1, fragB(vb + 32, hi), oacc[n]);
    }
  }

  // normalize by l (ones-column, dk col 96 -> lane hi*16 of tile 6),
  // merge heads into zattn[t][ch]
  float rinv[8];
#pragma unroll
  for (int j = 0; j < 8; ++j)
    rinv[j] = 1.0f / __shfl(oacc[6][j], hi << 4, 32);
#pragma unroll
  for (int n = 0; n < 6; ++n)
#pragma unroll
    for (int j = 0; j < 8; ++j)
      zattn[m0 + j + hi * 8][h2 * DK + n * 16 + lr] =
          (_Float16)(oacc[n][j] * rinv[j]);
  __syncthreads();

  // phase 3: out = Wo*attn + bo  (48 tiles, 6/wave)
  float* ob = out + (size_t)bidx * DMODEL * TSEQ;
  for (int tt = wave; tt < 48; tt += 8) {
    int mt = tt >> 2, nt = tt & 3;
    int row = mt * 16 + lr;
    v8f acc = {};
#pragma unroll
    for (int kc = 0; kc < 6; ++kc) {
      v16h a = fragA(wo16 + (size_t)row * DMODEL + kc * 32, hi);
      v16h b = fragB(&zattn[nt * 16 + lr][kc * 32], hi);
      acc = wmma16(a, b, acc);
    }
#pragma unroll
    for (int j = 0; j < 8; ++j) {
      int och = mt * 16 + j + hi * 8;
      ob[(size_t)och * TSEQ + t0 + nt * 16 + lr] = acc[j] + bo[och];
    }
  }
}

extern "C" void kernel_launch(void* const* d_in, const int* in_sizes, int n_in,
                              void* d_out, int out_size, void* d_ws,
                              size_t ws_size, hipStream_t stream) {
  (void)in_sizes; (void)n_in; (void)out_size; (void)ws_size;
  const float* z   = (const float*)d_in[0];
  const float* mem = (const float*)d_in[1];
  const float* Wq  = (const float*)d_in[2];
  const float* bq  = (const float*)d_in[3];
  const float* Wk  = (const float*)d_in[4];
  const float* bk  = (const float*)d_in[5];
  const float* Wv  = (const float*)d_in[6];
  const float* bv  = (const float*)d_in[7];
  const float* Wo  = (const float*)d_in[8];
  const float* bo  = (const float*)d_in[9];
  float* outp = (float*)d_out;

  _Float16* kh   = (_Float16*)d_ws;                      // [2][1024][96]
  _Float16* vT   = kh + (size_t)NH * BANKP * DK;         // [2][112][1024]
  _Float16* wq16 = vT + (size_t)NH * DKP * BANKP;        // [192][192]
  _Float16* wo16 = wq16 + (size_t)DMODEL * DMODEL;       // [192][192]

  kv_prep<<<(NH * DKP * BANKP) / 256, 256, 0, stream>>>(mem, Wk, bk, Wv, bv,
                                                        kh, vT);
  w_cvt<<<(DMODEL * DMODEL + 255) / 256, 256, 0, stream>>>(Wq, Wo, wq16, wo16);
  fused_attn<<<NB * (TSEQ / TT), 256, 0, stream>>>(z, wq16, bq, wo16, bo, kh,
                                                   vT, outp);
}
